// ACFModule_4217657885010
// MI455X (gfx1250) — matrix operand order repaired and checked by reading, not compiled, not run on hardware
//
#include <hip/hip_runtime.h>
#include <hip/hip_bf16.h>

// ---------------------------------------------------------------------------
// ACF attention module on gfx1250 (MI455X), wave32 + WMMA f16 16x16x32.
// Pipeline:
//   pool_pack   : avgpool3x3s2 -> pxt[b][n2][c] f16
//   pack_x      : x -> xt[b][n][c] f16
//   pack_w      : w_qs/w_vs/w_out -> f16 row-major
//   gemm_qkv    : WMMA GEMM 512x512 x N -> q_t[bh][n][dk], kt_t[bh][n2][dk],
//                 vt[bh][dv][n2] (all f16)
//   pi_kernel   : mixture gate softmax
//   attn_kernel : flash attention, 2 mixtures, online softmax, WMMA for
//                 scores + P*V, LDS round-trip for P layout conversion
//   gemm_out    : WMMA GEMM + BN + residual -> d_out f32
// ---------------------------------------------------------------------------

typedef __attribute__((ext_vector_type(16))) _Float16 v16h;
typedef __attribute__((ext_vector_type(8)))  _Float16 v8h;
typedef __attribute__((ext_vector_type(8)))  float    v8f;

#define NHEAD 8
#define DMODEL 512
#define NPIX 3136        // 56*56
#define NPIX2 784        // 28*28
#define BATCH 2

__device__ inline v16h mk16(v8h lo, v8h hi) {
    return __builtin_shufflevector(lo, hi, 0,1,2,3,4,5,6,7,8,9,10,11,12,13,14,15);
}
__device__ inline float rmax16(float v) {
    v = fmaxf(v, __shfl_xor(v, 1, 32));
    v = fmaxf(v, __shfl_xor(v, 2, 32));
    v = fmaxf(v, __shfl_xor(v, 4, 32));
    v = fmaxf(v, __shfl_xor(v, 8, 32));
    return v;
}
__device__ inline float rsum16(float v) {
    v += __shfl_xor(v, 1, 32);
    v += __shfl_xor(v, 2, 32);
    v += __shfl_xor(v, 4, 32);
    v += __shfl_xor(v, 8, 32);
    return v;
}

// ---- avgpool 3x3 s2 pad1 (count_include_pad=False) + transpose + f16 ------
__global__ void pool_pack(const float* __restrict__ x, _Float16* __restrict__ pxt) {
    int idx = blockIdx.x * blockDim.x + threadIdx.x;      // [0, 2*784*512)
    if (idx >= BATCH * NPIX2 * DMODEL) return;
    int c  = idx & 511;
    int n2 = (idx >> 9) % NPIX2;
    int b  = idx / (NPIX2 * DMODEL);
    int oh = n2 / 28, ow = n2 % 28;
    int h0 = oh * 2 - 1, w0 = ow * 2 - 1;
    float s = 0.0f; int cnt = 0;
    #pragma unroll
    for (int dh = 0; dh < 3; dh++)
        #pragma unroll
        for (int dw = 0; dw < 3; dw++) {
            int h = h0 + dh, w = w0 + dw;
            if (h >= 0 && h < 56 && w >= 0 && w < 56) {
                s += x[((size_t)(b * DMODEL + c) * 56 + h) * 56 + w];
                cnt++;
            }
        }
    pxt[idx] = (_Float16)(s / (float)cnt);
}

// ---- x [b][c][n] f32 -> xt [b][n][c] f16 ----------------------------------
__global__ void pack_x(const float* __restrict__ x, _Float16* __restrict__ xt) {
    int idx = blockIdx.x * blockDim.x + threadIdx.x;      // [0, 2*3136*512)
    if (idx >= BATCH * NPIX * DMODEL) return;
    int c = idx & 511;
    int n = (idx >> 9) % NPIX;
    int b = idx / (NPIX * DMODEL);
    xt[idx] = (_Float16)x[((size_t)(b * DMODEL + c) * NPIX) + n];
}

// ---- convert 3 weight matrices to f16 -------------------------------------
__global__ void pack_w(const float* __restrict__ wq, const float* __restrict__ wv,
                       const float* __restrict__ wo,
                       _Float16* __restrict__ wqh, _Float16* __restrict__ wvh,
                       _Float16* __restrict__ woh) {
    int idx = blockIdx.x * blockDim.x + threadIdx.x;      // [0, 3*262144)
    if (idx >= 3 * DMODEL * DMODEL) return;
    int which = idx / (DMODEL * DMODEL);
    int i = idx % (DMODEL * DMODEL);
    if (which == 0)      wqh[i] = (_Float16)wq[i];
    else if (which == 1) wvh[i] = (_Float16)wv[i];
    else                 woh[i] = (_Float16)wo[i];
}

// ---- generic 512x512 WMMA GEMM: C[o][n] = W[o][:] . X[b][n][:] + bias[o] --
// mode 0: dst = q_t[bh][n][dk]   (Npix = 3136)
// mode 1: dst = kt_t[bh][n2][dk] (Npix = 784)
// mode 2: dst = vt[bh][dv][n2]   (Npix = 784)
__global__ void gemm_qkv(const _Float16* __restrict__ W, const _Float16* __restrict__ X,
                         const float* __restrict__ bias, _Float16* __restrict__ dst,
                         int Npix, int mode) {
    int wave = threadIdx.x >> 5;
    int lane = threadIdx.x & 31;
    int hf   = lane >> 4;           // 0 or 1 (half of wave)
    int l16  = lane & 15;
    int mbase = blockIdx.x * 16;
    int ntile = blockIdx.y * 4 + wave;
    int b     = blockIdx.z;
    if (ntile >= (Npix >> 4)) return;
    int nbase = ntile * 16;

    const _Float16* Wrow = W + (size_t)(mbase + l16) * DMODEL;
    const _Float16* Xrow = X + ((size_t)b * Npix + nbase + l16) * DMODEL;

    v8f acc = {};
    for (int kk = 0; kk < DMODEL; kk += 32) {
        v8h a0 = *(const v8h*)(Wrow + kk + hf * 8);
        v8h a1 = *(const v8h*)(Wrow + kk + 16 + hf * 8);
        v8h b0 = *(const v8h*)(Xrow + kk + hf * 8);
        v8h b1 = *(const v8h*)(Xrow + kk + 16 + hf * 8);
        v16h af = mk16(a0, a1);
        v16h bf = mk16(b0, b1);
        acc = __builtin_amdgcn_wmma_f32_16x16x32_f16(false, af, false, bf,
                                                     (short)0, acc, false, false);
    }
    #pragma unroll
    for (int i = 0; i < 8; i++) {
        int o = mbase + i + 8 * hf;
        int n = nbase + l16;
        float v = acc[i] + bias[o];
        int head = o >> 6, dk = o & 63;
        size_t addr;
        if (mode == 2) addr = ((size_t)(b * NHEAD + head) * 64 + dk) * (size_t)Npix + n;
        else           addr = ((size_t)(b * NHEAD + head) * (size_t)Npix + n) * 64 + dk;
        dst[addr] = (_Float16)v;
    }
}

// ---- mixture gate: pi = softmax(mix_w @ mean_n(q)) ------------------------
__global__ void pi_kernel(const _Float16* __restrict__ Q, const float* __restrict__ mixw,
                          float* __restrict__ pi) {
    int bh = blockIdx.x;                 // 16
    int t = threadIdx.x;                 // 256
    int dk = t & 63, chunk = t >> 6;
    float s = 0.0f;
    for (int n = chunk; n < NPIX; n += 4)
        s += (float)Q[((size_t)bh * NPIX + n) * 64 + dk];
    __shared__ float tmp[256];
    tmp[t] = s;
    __syncthreads();
    if (t < 64) {
        float v = tmp[t] + tmp[t + 64] + tmp[t + 128] + tmp[t + 192];
        tmp[t] = v * (1.0f / (float)NPIX);
    }
    __syncthreads();
    if (t == 0) {
        float l0 = 0.0f, l1 = 0.0f;
        for (int d = 0; d < 64; d++) { l0 += mixw[d] * tmp[d]; l1 += mixw[64 + d] * tmp[d]; }
        float mx = fmaxf(l0, l1);
        float e0 = __expf(l0 - mx), e1 = __expf(l1 - mx);
        float inv = 1.0f / (e0 + e1);
        pi[bh * 2 + 0] = e0 * inv;
        pi[bh * 2 + 1] = e1 * inv;
    }
}

// ---- flash attention: 2 mixtures, online softmax, pi-weighted combine -----
__global__ void attn_kernel(const _Float16* __restrict__ Q,   // [bh][3136][64]
                            const _Float16* __restrict__ K,   // [bh][784][64]
                            const _Float16* __restrict__ V,   // [bh][64][784]
                            const float* __restrict__ pi,
                            _Float16* __restrict__ AO) {      // [b][3136][512]
    __shared__ __align__(16) _Float16 plds[4][2][16 * 32];    // per-wave P staging, 8KB
    int wave = threadIdx.x >> 5;
    int lane = threadIdx.x & 31;
    int hf   = lane >> 4;
    int l16  = lane & 15;
    int bh   = blockIdx.y;                  // 16
    int qbase = (blockIdx.x * 4 + wave) * 16;
    int b = bh >> 3, head = bh & 7;
    const float invT = 0.125f;              // 1/sqrt(D_K)

    // Q fragments for both mixtures (A layout: row = l16, K halves by hf)
    const _Float16* Qb = Q + ((size_t)bh * NPIX + qbase + l16) * 64;
    v16h qf[2];
    #pragma unroll
    for (int m = 0; m < 2; m++) {
        v8h lo = *(const v8h*)(Qb + m * 32 + hf * 16);
        v8h hi = *(const v8h*)(Qb + m * 32 + hf * 16 + 8);
        qf[m] = mk16(lo, hi);
    }

    v8f O[2][4];
    #pragma unroll
    for (int m = 0; m < 2; m++)
        #pragma unroll
        for (int j = 0; j < 4; j++) O[m][j] = (v8f){};
    float rm[2][8], rl[2][8];
    #pragma unroll
    for (int m = 0; m < 2; m++)
        #pragma unroll
        for (int i = 0; i < 8; i++) { rm[m][i] = -1e30f; rl[m][i] = 0.0f; }

    for (int kb = 0; kb < NPIX2; kb += 32) {
        bool v1ok = (kb + 16) < NPIX2;
        int kbb = v1ok ? kb + 16 : kb;      // clamped second sub-tile base

        // V fragments (B layout: col = dv, K = key), shared by both mixtures
        v16h vf[4];
        #pragma unroll
        for (int j = 0; j < 4; j++) {
            const _Float16* vp = V + ((size_t)bh * 64 + j * 16 + l16) * NPIX2;
            int koff = hf ? kbb : kb;
            v8h v0 = *(const v8h*)(vp + koff);
            v8h v1 = *(const v8h*)(vp + koff + 8);
            vf[j] = mk16(v0, v1);
        }

        #pragma unroll
        for (int m = 0; m < 2; m++) {
            // K fragments for the two 16-key sub-tiles
            const _Float16* kp0 = K + ((size_t)bh * NPIX2 + kb  + l16) * 64 + m * 32 + hf * 16;
            const _Float16* kp1 = K + ((size_t)bh * NPIX2 + kbb + l16) * 64 + m * 32 + hf * 16;
            v16h kf0 = mk16(*(const v8h*)kp0, *(const v8h*)(kp0 + 8));
            v16h kf1 = mk16(*(const v8h*)kp1, *(const v8h*)(kp1 + 8));

            v8f z = {};
            v8f s0 = __builtin_amdgcn_wmma_f32_16x16x32_f16(false, qf[m], false, kf0,
                                                            (short)0, z, false, false);
            v8f s1 = __builtin_amdgcn_wmma_f32_16x16x32_f16(false, qf[m], false, kf1,
                                                            (short)0, z, false, false);
            // scale + online softmax update (per output row)
            #pragma unroll
            for (int i = 0; i < 8; i++) {
                float a = s0[i] * invT;
                float c = v1ok ? s1[i] * invT : -1e30f;
                float tm = rmax16(fmaxf(a, c));
                float mn = fmaxf(rm[m][i], tm);
                float sc = __expf(rm[m][i] - mn);
                rm[m][i] = mn;
                float p0 = __expf(a - mn);
                float p1 = __expf(c - mn);
                s0[i] = p0; s1[i] = p1;
                rl[m][i] = rl[m][i] * sc + rsum16(p0 + p1);
                #pragma unroll
                for (int j = 0; j < 4; j++) O[m][j][i] *= sc;
            }
            // P (C-layout f32) -> A-fragment f16 via per-wave LDS slice
            _Float16* P = &plds[wave][m][0];
            #pragma unroll
            for (int i = 0; i < 8; i++) {
                int row = i + 8 * hf;
                P[row * 32 + l16]      = (_Float16)s0[i];
                P[row * 32 + 16 + l16] = (_Float16)s1[i];
            }
            asm volatile("s_wait_dscnt 0" ::: "memory");   // same-wave LDS RAW fence
            v8h plo = *(const v8h*)&plds[wave][m][l16 * 32 + hf * 16];
            v8h phi = *(const v8h*)&plds[wave][m][l16 * 32 + hf * 16 + 8];
            v16h pf = mk16(plo, phi);

            #pragma unroll
            for (int j = 0; j < 4; j++)
                O[m][j] = __builtin_amdgcn_wmma_f32_16x16x32_f16(false, pf, false, vf[j],
                                                                 (short)0, O[m][j],
                                                                 false, false);
        }
    }

    // combine mixtures with pi / l and write [b][n][head*64+dv] f16
    float p0 = pi[bh * 2 + 0], p1 = pi[bh * 2 + 1];
    #pragma unroll
    for (int i = 0; i < 8; i++) {
        float w0 = p0 / rl[0][i];
        float w1 = p1 / rl[1][i];
        int q = qbase + i + 8 * hf;
        #pragma unroll
        for (int j = 0; j < 4; j++) {
            float o = O[0][j][i] * w0 + O[1][j][i] * w1;
            AO[((size_t)(b * NPIX) + q) * DMODEL + head * 64 + j * 16 + l16] = (_Float16)o;
        }
    }
}

// ---- output projection + BN(eval) + residual ------------------------------
__global__ void gemm_out(const _Float16* __restrict__ Wo, const _Float16* __restrict__ AOt,
                         const float* __restrict__ gamma, const float* __restrict__ beta,
                         const float* __restrict__ x, float* __restrict__ out) {
    int wave = threadIdx.x >> 5;
    int lane = threadIdx.x & 31;
    int hf = lane >> 4, l16 = lane & 15;
    int mbase = blockIdx.x * 16;
    int nbase = (blockIdx.y * 4 + wave) * 16;
    int b = blockIdx.z;

    const _Float16* Wrow = Wo + (size_t)(mbase + l16) * DMODEL;
    const _Float16* Xrow = AOt + ((size_t)b * NPIX + nbase + l16) * DMODEL;
    v8f acc = {};
    for (int kk = 0; kk < DMODEL; kk += 32) {
        v8h a0 = *(const v8h*)(Wrow + kk + hf * 8);
        v8h a1 = *(const v8h*)(Wrow + kk + 16 + hf * 8);
        v8h b0 = *(const v8h*)(Xrow + kk + hf * 8);
        v8h b1 = *(const v8h*)(Xrow + kk + 16 + hf * 8);
        acc = __builtin_amdgcn_wmma_f32_16x16x32_f16(false, mk16(a0, a1), false,
                                                     mk16(b0, b1), (short)0, acc,
                                                     false, false);
    }
    const float inv = 0.999995000037f;     // 1/sqrt(1 + 1e-5)
    #pragma unroll
    for (int i = 0; i < 8; i++) {
        int co = mbase + i + 8 * hf;
        int n = nbase + l16;
        size_t a = ((size_t)(b * DMODEL + co) * NPIX) + n;
        out[a] = acc[i] * (gamma[co] * inv) + beta[co] + x[a];
    }
}

// ---------------------------------------------------------------------------
extern "C" void kernel_launch(void* const* d_in, const int* in_sizes, int n_in,
                              void* d_out, int out_size, void* d_ws, size_t ws_size,
                              hipStream_t stream) {
    const float* x     = (const float*)d_in[0];
    const float* w_qs  = (const float*)d_in[1];
    const float* b_qs  = (const float*)d_in[2];
    const float* w_vs  = (const float*)d_in[3];
    const float* b_vs  = (const float*)d_in[4];
    const float* mix_w = (const float*)d_in[5];
    const float* w_out = (const float*)d_in[6];
    const float* bn_g  = (const float*)d_in[7];
    const float* bn_b  = (const float*)d_in[8];
    float* out = (float*)d_out;

    // workspace carve (all 256B aligned)
    char* w = (char*)d_ws;
    auto take = [&](size_t bytes) { char* p = w; w += (bytes + 255) & ~(size_t)255; return p; };
    _Float16* xt   = (_Float16*)take((size_t)BATCH * NPIX  * DMODEL * 2);   // 6.4 MB
    _Float16* pxt  = (_Float16*)take((size_t)BATCH * NPIX2 * DMODEL * 2);   // 1.6 MB
    _Float16* wqh  = (_Float16*)take((size_t)DMODEL * DMODEL * 2);
    _Float16* wvh  = (_Float16*)take((size_t)DMODEL * DMODEL * 2);
    _Float16* woh  = (_Float16*)take((size_t)DMODEL * DMODEL * 2);
    _Float16* q_t  = (_Float16*)take((size_t)16 * NPIX  * 64 * 2);          // 6.4 MB
    _Float16* kt_t = (_Float16*)take((size_t)16 * NPIX2 * 64 * 2);
    _Float16* vt   = (_Float16*)take((size_t)16 * 64 * NPIX2 * 2);
    float*    piw  = (float*)take(16 * 2 * sizeof(float));
    _Float16* ao   = (_Float16*)take((size_t)BATCH * NPIX * DMODEL * 2);    // 6.4 MB

    pool_pack<<<dim3((BATCH * NPIX2 * DMODEL + 255) / 256), 256, 0, stream>>>(x, pxt);
    pack_x<<<dim3((BATCH * NPIX * DMODEL + 255) / 256), 256, 0, stream>>>(x, xt);
    pack_w<<<dim3((3 * DMODEL * DMODEL + 255) / 256), 256, 0, stream>>>(
        w_qs, w_vs, w_out, wqh, wvh, woh);

    gemm_qkv<<<dim3(32, 49, 2), 128, 0, stream>>>(wqh, xt,  b_qs, q_t,  NPIX,  0);
    gemm_qkv<<<dim3(32, 13, 2), 128, 0, stream>>>(wqh, pxt, b_qs, kt_t, NPIX2, 1);
    gemm_qkv<<<dim3(32, 13, 2), 128, 0, stream>>>(wvh, pxt, b_vs, vt,   NPIX2, 2);

    pi_kernel<<<dim3(16), 256, 0, stream>>>(q_t, mix_w, piw);

    attn_kernel<<<dim3(49, 16), 128, 0, stream>>>(q_t, kt_t, vt, piw, ao);

    gemm_out<<<dim3(32, 49, 2), 128, 0, stream>>>(woh, ao, bn_g, bn_b, x, out);
}
